// LengthTransformer_50259707298260
// MI455X (gfx1250) — matrix operand order repaired and checked by reading, not compiled, not run on hardware
//
#include <hip/hip_runtime.h>
#include <hip/hip_fp16.h>
#include <stdint.h>

// MI455X / gfx1250, wave32. Fused softmax-resample + GEMM using
// v_wmma_f32_16x16x32_f16. B=8, S=2048, D=512, L=2048 (from reference).
//
// 512-thread WG = 16 waves; tile LT=64 L-rows x full D=512.
// wave = (wl, wd): wl in {0,1} -> 2 L-subtiles of 16; wd in 0..7 -> 64 D-cols.
// Per-wave registers: acc 64 + staging 32 + frags 40 -> no VGPR-MSB thrash.
// Pipeline: [global loads c+1 in flight] [8x WMMA chunk c] [cvt_pk+LDS store c+1] barrier.

typedef __attribute__((ext_vector_type(16))) _Float16 v16h;
typedef __attribute__((ext_vector_type(8)))  float    v8f;

#define B_   8
#define S_   2048
#define D_   512
#define L_   2048
#define LT   64          // L rows per workgroup
#define KC   32          // WMMA K per chunk
#define NCH  (S_ / KC)   // 64 chunks
#define PPAD 40          // halfs per P row (32+8 pad): 80B rows
#define ZP   516         // dwords per Zpair row (512+4 pad) -> conflict-free frag reads
#define NT   512         // threads per block

union H8 { v16h h16; uint32_t u[8]; };

__global__ __launch_bounds__(NT)
void lt_fused_kernel(const float* __restrict__ z,
                     const float* __restrict__ zmask,
                     const int*   __restrict__ ls,
                     const float* __restrict__ sigma,
                     float*       __restrict__ out)
{
    __shared__ __align__(16) _Float16 Pbuf[2][LT][PPAD];    // A-matrix source (P chunk)
    __shared__ __align__(16) uint32_t Zpair[2][KC / 2][ZP]; // B source: packed {k,k+1} f16
    __shared__ float red[NT];
    __shared__ float invs[LT];
    __shared__ float nsh;

    const int tid    = threadIdx.x;
    const int lane   = tid & 31;
    const int wave   = tid >> 5;        // 0..15
    const int wd     = wave & 7;        // D-column group (64 cols)
    const int wl     = wave >> 3;       // L half: subtiles {0,1} or {2,3}
    const int half   = lane >> 4;
    const int lanelo = lane & 15;

    const int b     = blockIdx.x >> 5;          // 32 l-tiles per batch
    const int lbase = (blockIdx.x & 31) * LT;

    // ---- n_b = sum(z_mask[b,:])  (z_mask == (s < n)) ----
    {
        float s = 0.f;
        const float* zm = zmask + (size_t)b * S_;
        #pragma unroll
        for (int j = 0; j < 4; ++j) s += zm[tid * 4 + j];
        red[tid] = s;
        __syncthreads();
        for (int off = NT / 2; off > 0; off >>= 1) {
            if (tid < off) red[tid] += red[tid + off];
            __syncthreads();
        }
        if (tid == 0) nsh = red[0];
        __syncthreads();
    }
    const float nf     = nsh;
    const int   lsb    = ls[b];
    const float rate   = nf / (float)lsb;
    const float sg     = sigma[0];
    const float inv2s2 = 1.0f / (2.0f * sg * sg);
    const float LOG2E  = 1.44269504088896340736f;

    // P-producer role: 8 threads per row, 4 k's each
    const int   prow = tid >> 3;                 // 0..63
    const int   pkq  = tid & 7;                  // k offsets pkq*4 .. +3
    const float mu   = rate * (float)(lbase + prow);
    float rsum = 0.f;

    const float* zb = z + (size_t)b * S_ * D_;

    v8f acc[2][4];
    {
        v8f zero = {0.f,0.f,0.f,0.f,0.f,0.f,0.f,0.f};
        #pragma unroll
        for (int i = 0; i < 2; ++i)
            #pragma unroll
            for (int j = 0; j < 4; ++j) acc[i][j] = zero;
    }

    float4 zra[4], zrb[4];                       // in-flight staging registers

    // ---- issue 8 global b128 loads for chunk c (rows 2kp and 2kp+1) ----
    auto loadZ = [&](int c) {
        const int k0 = c * KC;
        #pragma unroll
        for (int u = 0; u < 4; ++u) {
            const int i  = u * NT + tid;         // 0..2047
            const int kp = i >> 7;               // 0..15
            const int d  = (i & 127) << 2;
            const float* p = zb + (size_t)(k0 + 2 * kp) * D_ + d;
            zra[u] = *(const float4*)(p);
            zrb[u] = *(const float4*)(p + D_);
        }
    };

    // ---- convert to packed f16 K-pairs (v_cvt_pk_f16_f32), b128 store ----
    auto storeZ = [&](int buf) {
        #pragma unroll
        for (int u = 0; u < 4; ++u) {
            const int i  = u * NT + tid;
            const int kp = i >> 7;
            const int d  = (i & 127) << 2;
            __half2 h0 = __floats2half2_rn(zra[u].x, zrb[u].x);
            __half2 h1 = __floats2half2_rn(zra[u].y, zrb[u].y);
            __half2 h2 = __floats2half2_rn(zra[u].z, zrb[u].z);
            __half2 h3 = __floats2half2_rn(zra[u].w, zrb[u].w);
            uint4 q;
            q.x = *(uint32_t*)&h0; q.y = *(uint32_t*)&h1;
            q.z = *(uint32_t*)&h2; q.w = *(uint32_t*)&h3;
            *(uint4*)(&Zpair[buf][kp][d]) = q;
        }
    };

    // ---- P[l, k0..k0+31] -> exp(logits), f16, plus running row-sum partials ----
    auto stageP = [&](int c, int buf) {
        const int k0 = c * KC;
        float p4[4];
        #pragma unroll
        for (int j = 0; j < 4; ++j) {
            const float sidx = (float)(k0 + pkq * 4 + j);
            float d = sidx - mu;
            d = fminf(fmaxf(d, -100.f), 100.f);
            const float e = exp2f(d);                       // 2^distance
            float p = exp2f(-e * inv2s2 * LOG2E);           // exp(logit)
            p = (sidx < nf) ? p : 0.f;                      // mask (exp(-99) ~ 0)
            rsum += p;
            p4[j] = p;
        }
        __half2 h0 = __floats2half2_rn(p4[0], p4[1]);
        __half2 h1 = __floats2half2_rn(p4[2], p4[3]);
        uint2 q; q.x = *(uint32_t*)&h0; q.y = *(uint32_t*)&h1;
        *(uint2*)(&Pbuf[buf][prow][pkq * 4]) = q;           // b64 LDS store
    };

    auto compute = [&](int buf) {
        // B fragments: 32x16 f16, lanes0-15 K=0..15, lanes16-31 K=16..31.
        // Packed-pair layout: VGPR v = Zpair[half*8 + v][col]  (one ds_load_b32)
        H8 bf[4];
        #pragma unroll
        for (int nd = 0; nd < 4; ++nd) {
            const int col = wd * 64 + nd * 16 + lanelo;
            #pragma unroll
            for (int v = 0; v < 8; ++v)
                bf[nd].u[v] = Zpair[buf][half * 8 + v][col];
        }
        #pragma unroll
        for (int la = 0; la < 2; ++la) {
            // A fragment: 16x32 f16 (lanes0-15: K 0-15; lanes16-31: K 16-31)
            H8 af;
            const int row = (wl * 2 + la) * 16 + lanelo;
            #pragma unroll
            for (int v = 0; v < 8; ++v) {
                const int kb = half * 16 + (v << 1);
                af.u[v] = *(const uint32_t*)&Pbuf[buf][row][kb];
            }
            #pragma unroll
            for (int nd = 0; nd < 4; ++nd) {
                acc[la][nd] = __builtin_amdgcn_wmma_f32_16x16x32_f16(
                    false, af.h16, false, bf[nd].h16,
                    (short)0, acc[la][nd], false, false);
            }
        }
    };

    // Prologue: stage chunk 0
    loadZ(0);
    storeZ(0);
    stageP(0, 0);
    __syncthreads();

    for (int c = 0; c < NCH; ++c) {
        const int buf = c & 1;
        if (c + 1 < NCH) loadZ(c + 1);      // global loads in flight during WMMAs
        compute(buf);                       // ds reads of buf + 8x v_wmma
        if (c + 1 < NCH) {
            storeZ(buf ^ 1);                // waits on loads land here, after compute
            stageP(c + 1, buf ^ 1);
        }
        __syncthreads();
    }

    // ---- row sums -> reciprocal, plus z_prime_mask tail output ----
    red[tid] = rsum;
    __syncthreads();
    if (tid < LT) {
        float s = 0.f;
        #pragma unroll
        for (int j = 0; j < 8; ++j) s += red[8 * tid + j];
        const float m = ((lbase + tid) < lsb) ? 1.f : 0.f;
        invs[tid] = m / s;                  // mask folded into scale
        out[(size_t)B_ * L_ * D_ + (size_t)b * L_ + lbase + tid] = m;
    }
    __syncthreads();

    // ---- write z' (C/D layout: lanes0-15 -> M=v, lanes16-31 -> M=8+v) ----
    float* ob = out + (size_t)b * L_ * D_;
    #pragma unroll
    for (int la = 0; la < 2; ++la) {
        #pragma unroll
        for (int nd = 0; nd < 4; ++nd) {
            const int col = wd * 64 + nd * 16 + lanelo;
            #pragma unroll
            for (int v = 0; v < 8; ++v) {
                const int lloc = (wl * 2 + la) * 16 + half * 8 + v;
                ob[(size_t)(lbase + lloc) * D_ + col] = acc[la][nd][v] * invs[lloc];
            }
        }
    }
}

extern "C" void kernel_launch(void* const* d_in, const int* in_sizes, int n_in,
                              void* d_out, int out_size, void* d_ws, size_t ws_size,
                              hipStream_t stream) {
    const float* z   = (const float*)d_in[0];
    const float* zm  = (const float*)d_in[1];
    const int*   ls  = (const int*)d_in[2];
    const float* sg  = (const float*)d_in[3];
    float*       out = (float*)d_out;

    dim3 grid(B_ * (L_ / LT));   // 256 workgroups
    dim3 block(NT);              // 16 waves (wave32)
    hipLaunchKernelGGL(lt_fused_kernel, grid, block, 0, stream, z, zm, ls, sg, out);
}
